// ShallowRBF_64226940944944
// MI455X (gfx1250) — matrix operand order repaired and checked by reading, not compiled, not run on hardware
//
#include <hip/hip_runtime.h>
#include <hip/hip_bf16.h>

#define Bsz  8192
#define Dd   2048
#define Cc   4096
#define Kk   1000
#define KPAD 1024

typedef __attribute__((ext_vector_type(16))) __bf16 v16bf;
typedef __attribute__((ext_vector_type(8)))  __bf16 v8bf;
typedef __attribute__((ext_vector_type(8)))  float  v8f;
typedef __attribute__((ext_vector_type(4)))  unsigned int u32x4;
typedef __attribute__((ext_vector_type(8)))  int i32x8;
typedef __attribute__((ext_vector_type(4)))  int i32x4;

#if defined(__HIP_DEVICE_COMPILE__) && defined(__has_builtin)
#if __has_builtin(__builtin_amdgcn_tensor_load_to_lds) && __has_builtin(__builtin_amdgcn_s_wait_tensorcnt)
#define HAVE_TDM 1
#endif
#endif
#ifndef HAVE_TDM
#define HAVE_TDM 0
#endif

__device__ __forceinline__ __bf16 f2bf(float f) {
    unsigned u = __builtin_bit_cast(unsigned, f);
    unsigned r = u + 0x7FFFu + ((u >> 16) & 1u);           // round-to-nearest-even
    unsigned short h = (unsigned short)(r >> 16);
    return __builtin_bit_cast(__bf16, h);
}

__device__ __forceinline__ v16bf mk16(v8bf lo, v8bf hi) {
    v16bf r;
#pragma unroll
    for (int i = 0; i < 8; ++i) { r[i] = lo[i]; r[i + 8] = hi[i]; }
    return r;
}

#if HAVE_TDM
// TDM: load a 128(center)x64(dim) bf16 tile from row-major Cbf(4096x2048) into LDS,
// padding each 128-byte row with 16 bytes -> 144-byte LDS row pitch (== cs[..][72]).
__device__ __forceinline__ void tdm_load_centers(const __bf16* gsrc, unsigned ldsAddr) {
    unsigned long long ga = (unsigned long long)(uintptr_t)gsrc;
    u32x4 g0;
    g0.x = 1u;                                             // count=1, user descriptor
    g0.y = ldsAddr;                                        // LDS byte address
    g0.z = (unsigned)ga;                                   // global addr lo
    g0.w = (unsigned)((ga >> 32) & 0x1FFFFFFull) | (2u << 30);  // addr hi | type=2
    i32x8 g1;
    // data_size=1 (2B) | pad_enable | pad_interval=4 (32 DW = 128B) | pad_amount=3 (4 DW = 16B)
    g1[0] = (int)((1u << 16) | (1u << 20) | (4u << 22) | (3u << 25));
    g1[1] = (int)((unsigned)Dd << 16);                     // tensor_dim0 = 2048
    g1[2] = (int)((unsigned)Cc << 16);                     // tensor_dim1 = 4096
    g1[3] = (int)(64u << 16);                              // tile_dim0 = 64 elems
    g1[4] = 128;                                           // tile_dim1 = 128 rows
    g1[5] = Dd;                                            // tensor_dim0_stride = 2048
    g1[6] = 0;
    g1[7] = 0;
    i32x4 z4 = {0, 0, 0, 0};
#if __clang_major__ >= 23
    i32x8 z8 = {0, 0, 0, 0, 0, 0, 0, 0};
    __builtin_amdgcn_tensor_load_to_lds(g0, g1, z4, z4, z8, 0);
#else
    __builtin_amdgcn_tensor_load_to_lds(g0, g1, z4, z4, 0);
#endif
}
#endif

// ---- one-time prep: W (1000x4096 f32) -> Wbf (1024x4096 bf16, zero-padded rows) ----
__global__ __launch_bounds__(256) void prep_w(const float* __restrict__ W,
                                              __bf16* __restrict__ Wbf) {
    size_t base = ((size_t)blockIdx.x * 256 + threadIdx.x) * 8;
    size_t row = base >> 12;  // /4096
    v8bf r;
    if (row < Kk) {
        float4 a = *reinterpret_cast<const float4*>(W + base);
        float4 c = *reinterpret_cast<const float4*>(W + base + 4);
        r[0] = f2bf(a.x); r[1] = f2bf(a.y); r[2] = f2bf(a.z); r[3] = f2bf(a.w);
        r[4] = f2bf(c.x); r[5] = f2bf(c.y); r[6] = f2bf(c.z); r[7] = f2bf(c.w);
    } else {
        __bf16 z = __builtin_bit_cast(__bf16, (unsigned short)0);
#pragma unroll
        for (int i = 0; i < 8; ++i) r[i] = z;
    }
    *reinterpret_cast<v8bf*>(Wbf + base) = r;
}

// ---- one-time prep: centers -> bf16 + f32 squared norms ----
__global__ __launch_bounds__(256) void prep_c(const float* __restrict__ C,
                                              __bf16* __restrict__ Cbf,
                                              float* __restrict__ c2) {
    __shared__ float wsum[8];
    int row = blockIdx.x, t = threadIdx.x;
    size_t base = (size_t)row * Dd + t * 8;
    float4 a = *reinterpret_cast<const float4*>(C + base);
    float4 c = *reinterpret_cast<const float4*>(C + base + 4);
    v8bf r;
    r[0] = f2bf(a.x); r[1] = f2bf(a.y); r[2] = f2bf(a.z); r[3] = f2bf(a.w);
    r[4] = f2bf(c.x); r[5] = f2bf(c.y); r[6] = f2bf(c.z); r[7] = f2bf(c.w);
    *reinterpret_cast<v8bf*>(Cbf + base) = r;
    float s = a.x * a.x + a.y * a.y + a.z * a.z + a.w * a.w +
              c.x * c.x + c.y * c.y + c.z * c.z + c.w * c.w;
    s += __shfl_xor(s, 1, 32);  s += __shfl_xor(s, 2, 32);
    s += __shfl_xor(s, 4, 32);  s += __shfl_xor(s, 8, 32);
    s += __shfl_xor(s, 16, 32);
    if ((t & 31) == 0) wsum[t >> 5] = s;
    __syncthreads();
    if (t == 0) {
        float tot = 0.f;
#pragma unroll
        for (int i = 0; i < 8; ++i) tot += wsum[i];
        c2[row] = tot;
    }
}

// One workgroup = 16 rows of x, 8 wave32s.
// GEMM1: each wave owns 16 centers of the current 128-center chunk (TDM-staged, double-buffered).
// GEMM2: each wave owns 128 output classes (8 WMMA n-tiles), accumulators persist.
__global__ __launch_bounds__(256, 2)
void rbf_fused(const float* __restrict__ x, const __bf16* __restrict__ Cbf,
               const float* __restrict__ beta, const __bf16* __restrict__ Wbf,
               const float* __restrict__ bias, const float* __restrict__ c2g,
               float* __restrict__ out) {
    __shared__ __align__(16) __bf16 xs[16][72];        // x chunk 16x64 (144B rows)
    __shared__ __align__(16) __bf16 cs[2][128][72];    // centers chunk, double buffered
    __shared__ __align__(16) __bf16 acts[16][136];     // act chunk 16x128 (272B rows)
    __shared__ float x2s[16];
    __shared__ float rsum[16];

    const int tid      = threadIdx.x;
    const int wave     = tid >> 5;     // wave32
    const int lane     = tid & 31;
    const int lanelo   = lane & 15;
    const int lanehalf = lane >> 4;
    const int rowBase  = blockIdx.x * 16;

#if HAVE_TDM
    unsigned csAddr[2];
    csAddr[0] = (unsigned)(unsigned long long)(uintptr_t)&cs[0][0][0];
    csAddr[1] = (unsigned)(unsigned long long)(uintptr_t)&cs[1][0][0];
#endif

    if (tid < 16) { x2s[tid] = 0.f; rsum[tid] = 0.f; }
    __syncthreads();

    // ---- x row norms (f32, exact) ----
    {
        int r = tid >> 4, seg = tid & 15;
        const float* xp = x + (size_t)(rowBase + r) * Dd + seg * 128;
        float s = 0.f;
#pragma unroll 8
        for (int i = 0; i < 128; i += 4) {
            float4 v = *reinterpret_cast<const float4*>(xp + i);
            s += v.x * v.x + v.y * v.y + v.z * v.z + v.w * v.w;
        }
        atomicAdd(&x2s[r], s);
    }

    v8f acc2[8];
#pragma unroll
    for (int nt = 0; nt < 8; ++nt) acc2[nt] = (v8f){0.f, 0.f, 0.f, 0.f, 0.f, 0.f, 0.f, 0.f};

    for (int cc = 0; cc < 32; ++cc) {
        const int cbase = cc * 128;
        v8f acc1 = (v8f){0.f, 0.f, 0.f, 0.f, 0.f, 0.f, 0.f, 0.f};

#if HAVE_TDM
        if (tid < 32) tdm_load_centers(Cbf + (size_t)cbase * Dd, csAddr[0]);
#endif
        for (int dc = 0; dc < 32; ++dc) {
            const int dbase = dc * 64;
            const int buf = dc & 1;
            __syncthreads();  // previous readers of xs / cs[buf^1] are done
#if HAVE_TDM
            if (tid < 32 && dc + 1 < 32)
                tdm_load_centers(Cbf + (size_t)cbase * Dd + (dc + 1) * 64, csAddr[(dc + 1) & 1]);
#else
            {   // fallback: cooperative bf16 copy, 64B per thread
                int cl = tid >> 1, d0 = (tid & 1) * 32;
                const v8bf* src = reinterpret_cast<const v8bf*>(
                    Cbf + (size_t)(cbase + cl) * Dd + dbase + d0);
                v8bf* dst = reinterpret_cast<v8bf*>(&cs[buf][cl][d0]);
                dst[0] = src[0]; dst[1] = src[1]; dst[2] = src[2]; dst[3] = src[3];
            }
#endif
            // stage x chunk: thread -> (row, 4 dims), f32 -> bf16 inline (tiny)
            {
                int r = tid >> 4, d0 = (tid & 15) * 4;
                const float* xp = x + (size_t)(rowBase + r) * Dd + dbase + d0;
                if (dc + 1 < 32) __builtin_prefetch(xp + 64, 0, 1);  // global_prefetch_b8
                float4 v = *reinterpret_cast<const float4*>(xp);
                xs[r][d0 + 0] = f2bf(v.x); xs[r][d0 + 1] = f2bf(v.y);
                xs[r][d0 + 2] = f2bf(v.z); xs[r][d0 + 3] = f2bf(v.w);
            }
#if HAVE_TDM
            if (tid < 32) {
                if (dc + 1 < 32) __builtin_amdgcn_s_wait_tensorcnt(1);  // buf ready, next in flight
                else             __builtin_amdgcn_s_wait_tensorcnt(0);
            }
#endif
            __syncthreads();
            // GEMM1: two 16x16x32 bf16 WMMAs per D-chunk
#pragma unroll
            for (int ks = 0; ks < 2; ++ks) {
                int ka = ks * 32 + lanehalf * 16;
                v8bf alo = *reinterpret_cast<const v8bf*>(&xs[lanelo][ka]);
                v8bf ahi = *reinterpret_cast<const v8bf*>(&xs[lanelo][ka + 8]);
                v16bf A = mk16(alo, ahi);
                int clc = wave * 16 + lanelo;
                int kb = ks * 32 + lanehalf * 16;
                v8bf blo = *reinterpret_cast<const v8bf*>(&cs[buf][clc][kb]);
                v8bf bhi = *reinterpret_cast<const v8bf*>(&cs[buf][clc][kb + 8]);
                v16bf Bm = mk16(blo, bhi);
                acc1 = __builtin_amdgcn_wmma_f32_16x16x32_bf16(
                    false, A, false, Bm, (short)0, acc1, false, false);
            }
        }

        // ---- RBF epilogue: sq -> dist -> exp; write act tile; rowsum ----
        {
            int clc = wave * 16 + lanelo;
            float c2 = c2g[cbase + clc];
            float bv = beta[cbase + clc];
            float av[8];
#pragma unroll
            for (int r = 0; r < 8; ++r) {
                int m = r + lanehalf * 8;
                float sq = x2s[m] + c2 - 2.f * acc1[r];
                sq = fmaxf(sq, 0.f);
                float a = __expf(-bv * sqrtf(sq));
                av[r] = a;
                acts[m][clc] = f2bf(a);
            }
#pragma unroll
            for (int r = 0; r < 8; ++r) {  // reduce over this wave's 16 columns
                float v = av[r];
                v += __shfl_xor(v, 1, 32);
                v += __shfl_xor(v, 2, 32);
                v += __shfl_xor(v, 4, 32);
                v += __shfl_xor(v, 8, 32);
                if (lanelo == 0) atomicAdd(&rsum[r + lanehalf * 8], v);
            }
        }
        __syncthreads();

        // ---- GEMM2: act(16x128) @ Wbf^T chunk -> persistent accumulators ----
#pragma unroll
        for (int kc = 0; kc < 4; ++kc) {
            int ka = kc * 32 + lanehalf * 16;
            v8bf alo = *reinterpret_cast<const v8bf*>(&acts[lanelo][ka]);
            v8bf ahi = *reinterpret_cast<const v8bf*>(&acts[lanelo][ka + 8]);
            v16bf A = mk16(alo, ahi);
#pragma unroll
            for (int nt = 0; nt < 8; ++nt) {
                int col = wave * 128 + nt * 16 + lanelo;      // 0..1023, Wbf zero-padded
                const __bf16* wp = Wbf + (size_t)col * Cc + cbase + kc * 32 + lanehalf * 16;
                v8bf blo = *reinterpret_cast<const v8bf*>(wp);
                v8bf bhi = *reinterpret_cast<const v8bf*>(wp + 8);
                v16bf Bm = mk16(blo, bhi);
                acc2[nt] = __builtin_amdgcn_wmma_f32_16x16x32_bf16(
                    false, A, false, Bm, (short)0, acc2[nt], false, false);
            }
        }
    }

    __syncthreads();  // rsum final
    float rinv[8];
#pragma unroll
    for (int r = 0; r < 8; ++r) rinv[r] = 1.f / rsum[r + lanehalf * 8];
#pragma unroll
    for (int nt = 0; nt < 8; ++nt) {
        int col = wave * 128 + nt * 16 + lanelo;
        if (col < Kk) {
            float bb = bias[col];
#pragma unroll
            for (int r = 0; r < 8; ++r) {
                int m = r + lanehalf * 8;
                out[(size_t)(rowBase + m) * Kk + col] = acc2[nt][r] * rinv[r] + bb;
            }
        }
    }
}

extern "C" void kernel_launch(void* const* d_in, const int* in_sizes, int n_in,
                              void* d_out, int out_size, void* d_ws, size_t ws_size,
                              hipStream_t stream) {
    const float* x       = (const float*)d_in[0];
    const float* centers = (const float*)d_in[1];
    const float* beta    = (const float*)d_in[2];
    const float* W       = (const float*)d_in[3];
    const float* b       = (const float*)d_in[4];
    float* out = (float*)d_out;
    (void)in_sizes; (void)n_in; (void)out_size; (void)ws_size;

    // ws layout: Wbf (1024x4096 bf16) | Cbf (4096x2048 bf16) | c2 (4096 f32)  ~= 25.2 MB
    __bf16* Wbf = (__bf16*)d_ws;
    __bf16* Cbf = (__bf16*)((char*)d_ws + (size_t)KPAD * Cc * 2);
    float*  c2g = (float*)((char*)d_ws + (size_t)KPAD * Cc * 2 + (size_t)Cc * Dd * 2);

    prep_w<<<dim3((KPAD * Cc) / (256 * 8)), dim3(256), 0, stream>>>(W, Wbf);
    prep_c<<<dim3(Cc), dim3(256), 0, stream>>>(centers, Cbf, c2g);
    rbf_fused<<<dim3(Bsz / 16), dim3(256), 0, stream>>>(x, Cbf, beta, Wbf, b, c2g, out);
}
